// WindowAttention_15461882266041
// MI455X (gfx1250) — matrix operand order repaired and checked by reading, not compiled, not run on hardware
//
#include <hip/hip_runtime.h>
#include <hip/hip_bf16.h>
#include <stdint.h>

typedef __bf16 bf16x16 __attribute__((ext_vector_type(16)));
typedef float  floatx8 __attribute__((ext_vector_type(8)));
typedef unsigned short u16;

#define NTOK   49
#define DIMC   128
#define NH     4
#define HD     32
#define NWIN   64
#define SCALEQ 0.17677669529663687f

// LDS strides (in u16 elements)
#define LDX 136   // xbf / Q / K / O : 64 rows x 136
#define LDV 72    // Vt : 128 rows x 72
#define LDP 72    // P  : per head 64 rows x 72

// LDS layout (bytes). Aliasing: [xf32|xbf] -> P after phase1 ; Q -> O after S-phase.
#define OFF_XF32 0        // 49*128*4  = 25088
#define OFF_XBF  25088    // 64*136*2  = 17408   (P alias region = [0, 42496), P needs 36864)
#define OFF_Q    42496    // 17408   (O alias)
#define OFF_K    59904    // 17408
#define OFF_VT   77312    // 128*72*2 = 18432
#define SMEM_BYTES 95744

__device__ __forceinline__ u16 f2bf(float f) {
  unsigned u = __float_as_uint(f);
  unsigned r = u + 0x7FFFu + ((u >> 16) & 1u);
  return (u16)(r >> 16);
}

__device__ __forceinline__ uint4 pack8bf(floatx8 d) {
  uint4 q;
  q.x = (unsigned)f2bf(d[0]) | ((unsigned)f2bf(d[1]) << 16);
  q.y = (unsigned)f2bf(d[2]) | ((unsigned)f2bf(d[3]) << 16);
  q.z = (unsigned)f2bf(d[4]) | ((unsigned)f2bf(d[5]) << 16);
  q.w = (unsigned)f2bf(d[6]) | ((unsigned)f2bf(d[7]) << 16);
  return q;
}

union FragU { uint4 q[2]; bf16x16 v; };

// A-fragment (16x32 bf16) from row-major [ld] at (m0,k0).
// Lane L<16 holds row m0+L, K = {k0..k0+7} U {k0+16..k0+23}; lane L+16 the other halves.
__device__ __forceinline__ bf16x16 ldA(const u16* base, int ld, int m0, int k0, int lane) {
  int r = lane & 15, h = lane >> 4;
  const u16* p = base + (m0 + r) * ld + k0 + h * 8;
  FragU u;
  u.q[0] = *(const uint4*)p;
  u.q[1] = *(const uint4*)(p + 16);
  return u.v;
}

// B-fragment (32x16 bf16) where Bt[n][k] = B[k][n] is row-major with stride ld.
// Lane L<16 holds column n0+L, K = {k0..k0+7} U {k0+16..k0+23}; lane L+16 the other halves.
__device__ __forceinline__ bf16x16 ldB(const u16* baseT, int ld, int n0, int k0, int lane) {
  int r = lane & 15, h = lane >> 4;
  const u16* p = baseT + (n0 + r) * ld + k0 + h * 8;
  FragU u;
  u.q[0] = *(const uint4*)p;
  u.q[1] = *(const uint4*)(p + 16);
  return u.v;
}

__device__ __forceinline__ floatx8 wmma_bf16(bf16x16 a, bf16x16 b, floatx8 c) {
  return __builtin_amdgcn_wmma_f32_16x16x32_bf16(false, a, false, b, (short)0, c, false, false);
}

// ---------------------------------------------------------------------------
// Prep kernel: fp32 -> bf16 weights; gather relative-position bias to (nH,49,49)
// ---------------------------------------------------------------------------
__global__ void swin_prep_kernel(const float* __restrict__ qkv_w,
                                 const float* __restrict__ proj_w,
                                 const float* __restrict__ bias_table,
                                 const int*   __restrict__ rel_index,
                                 u16* __restrict__ qkvw_bf,
                                 u16* __restrict__ projw_bf,
                                 float* __restrict__ biasPre) {
  int i = blockIdx.x * blockDim.x + threadIdx.x;
  const int n1 = 384 * DIMC, n2 = DIMC * DIMC, n3 = NH * NTOK * NTOK;
  if (i < n1) {
    qkvw_bf[i] = f2bf(qkv_w[i]);
  } else if (i < n1 + n2) {
    int j = i - n1;
    projw_bf[j] = f2bf(proj_w[j]);
  } else if (i < n1 + n2 + n3) {
    int j = i - n1 - n2;
    int hd = j / (NTOK * NTOK);
    int mn = j - hd * (NTOK * NTOK);
    biasPre[j] = bias_table[rel_index[mn] * NH + hd];
  }
}

// ---------------------------------------------------------------------------
// Fused window-attention kernel: one workgroup (8 wave32) per window.
// ---------------------------------------------------------------------------
__global__ __launch_bounds__(256)
void swin_attn_kernel(const float* __restrict__ x, const float* __restrict__ mask,
                      const float* __restrict__ qkv_b, const float* __restrict__ proj_b,
                      const u16* __restrict__ qkvw, const u16* __restrict__ projw,
                      const float* __restrict__ biasPre, float* __restrict__ out) {
  __shared__ __align__(16) unsigned char smem[SMEM_BYTES];
  float* xf32 = (float*)(smem + OFF_XF32);
  u16*   xbf  = (u16*)(smem + OFF_XBF);
  u16*   Pm   = (u16*)(smem + 0);        // alias of [xf32|xbf] after phase 1
  u16*   Qm   = (u16*)(smem + OFF_Q);
  u16*   Om   = (u16*)(smem + OFF_Q);    // alias of Q after S-phase
  u16*   Km   = (u16*)(smem + OFF_K);
  u16*   Vt   = (u16*)(smem + OFF_VT);

  const int tid  = threadIdx.x;
  const int lane = tid & 31;
  const int wid  = tid >> 5;
  const int r16  = lane & 15;
  const int hh   = lane >> 4;
  const int blk  = blockIdx.x;
  const int wm   = blk & (NWIN - 1);

  // Warm L2/WGP$ for the shared bf16 weights (global_prefetch_b8).
  __builtin_prefetch(qkvw + wid * 4096, 0, 3);
  __builtin_prefetch(projw + wid * 2048, 0, 3);

  // ---- Phase 0: async-copy x window (49x128 fp32) to LDS, convert to bf16 ----
  {
    const float* xw = x + (size_t)blk * (NTOK * DIMC);
    unsigned long long sbase = (unsigned long long)(uintptr_t)xw;
    for (int i = tid; i < (NTOK * DIMC) / 4; i += 256) {
      unsigned ldsOff = OFF_XF32 + (unsigned)i * 16u;  // static LDS block starts at 0
      unsigned vOff   = (unsigned)i * 16u;
      asm volatile("global_load_async_to_lds_b128 %0, %1, %2 offset:0"
                   :: "v"(ldsOff), "v"(vOff), "s"(sbase) : "memory");
    }
    asm volatile("s_wait_asynccnt 0" ::: "memory");
    __syncthreads();
#pragma unroll 4
    for (int i = tid; i < NTOK * DIMC; i += 256)
      xbf[(i >> 7) * LDX + (i & 127)] = f2bf(xf32[i]);
    for (int i = tid; i < (64 - NTOK) * DIMC; i += 256)       // zero token padding rows
      xbf[(NTOK + (i >> 7)) * LDX + (i & 127)] = 0;
    __syncthreads();
  }

  // ---- Phase 1a: qkv^T = Wqk @ x^T  -> Q,K row-major [token][feat] bf16 ----
  // 64 tiles (16 featTiles x 4 tokTiles), 8 per wave, K=128 (4 wmma each).
  for (int i = 0; i < 8; ++i) {
    int t = wid + 8 * i;
    int mt = t >> 2, nt = t & 3;
    floatx8 acc = {};
#pragma unroll
    for (int kk = 0; kk < 4; ++kk) {
      bf16x16 A = ldA(qkvw, DIMC, mt * 16, kk * 32, lane);    // W rows (contig)
      bf16x16 B = ldB(xbf, LDX, nt * 16, kk * 32, lane);      // x rows (contig)
      acc = wmma_bf16(A, B, acc);
    }
    int token = nt * 16 + r16;
    int featG = mt * 16 + 8 * hh;
    floatx8 wb;
#pragma unroll
    for (int r = 0; r < 8; ++r) wb[r] = acc[r] + qkv_b[featG + r];
    u16* dst = ((mt < 8) ? Qm : Km) + token * LDX + (mt & 7) * 16 + 8 * hh;
    *(uint4*)dst = pack8bf(wb);                               // contiguous row store
  }

  // ---- Phase 1b: V = x @ Wv^T -> Vt transposed [dim][token] bf16 ----
  for (int i = 0; i < 4; ++i) {
    int t = wid + 8 * i;
    int mt = t & 3, nt = t >> 2;                              // 4 tokTiles x 8 dimTiles
    floatx8 acc = {};
#pragma unroll
    for (int kk = 0; kk < 4; ++kk) {
      bf16x16 A = ldA(xbf, LDX, mt * 16, kk * 32, lane);
      bf16x16 B = ldB(qkvw + 256 * DIMC, DIMC, nt * 16, kk * 32, lane);
      acc = wmma_bf16(A, B, acc);
    }
    int dim = nt * 16 + r16;
    float bv = qkv_b[256 + dim];
    floatx8 wb;
#pragma unroll
    for (int r = 0; r < 8; ++r) wb[r] = acc[r] + bv;
    *(uint4*)(Vt + dim * LDV + mt * 16 + 8 * hh) = pack8bf(wb);   // contiguous (tokens)
  }
  __syncthreads();

  // ---- Phase 2: S = Q K^T * scale + bias + mask ; in-register softmax -> P bf16 ----
  // 16 strips (head, m-tile); each wave owns a full 16-row strip (all 4 key tiles).
  for (int si = 0; si < 2; ++si) {
    int s = wid + 8 * si;
    int hd = s >> 2;
    int m0 = (s & 3) * 16;
    bf16x16 A = ldA(Qm, LDX, m0, hd * HD, lane);
    floatx8 accs[4];
#pragma unroll
    for (int t4 = 0; t4 < 4; ++t4)
      accs[t4] = wmma_bf16(A, ldB(Km, LDX, t4 * 16, hd * HD, lane), (floatx8){});

    const float* bp = biasPre + hd * (NTOK * NTOK);
    const float* mp = mask + (size_t)wm * (NTOK * NTOK);
#pragma unroll
    for (int t4 = 0; t4 < 4; ++t4) {
#pragma unroll
      for (int r = 0; r < 8; ++r) {
        int m = m0 + 8 * hh + r;
        int n = t4 * 16 + r16;
        float v = -30.0f;                                     // pad keys/rows
        if (m < NTOK && n < NTOK)
          v = accs[t4][r] * SCALEQ + bp[m * NTOK + n] + mp[m * NTOK + n];
        accs[t4][r] = v;
      }
    }
    u16* Ph = Pm + hd * (64 * LDP);
#pragma unroll
    for (int r = 0; r < 8; ++r) {
      // Row m = m0+8*hh+r lives in the 16 lanes of this half-wave.
      float mx = fmaxf(fmaxf(accs[0][r], accs[1][r]), fmaxf(accs[2][r], accs[3][r]));
#pragma unroll
      for (int d = 8; d >= 1; d >>= 1) mx = fmaxf(mx, __shfl_xor(mx, d, 32));
      float p0 = __expf(accs[0][r] - mx), p1 = __expf(accs[1][r] - mx);
      float p2 = __expf(accs[2][r] - mx), p3 = __expf(accs[3][r] - mx);
      float sum = p0 + p1 + p2 + p3;
#pragma unroll
      for (int d = 8; d >= 1; d >>= 1) sum += __shfl_xor(sum, d, 32);
      float inv = 1.0f / sum;
      u16* prow = Ph + (m0 + 8 * hh + r) * LDP + r16;
      prow[0]  = f2bf(p0 * inv);
      prow[16] = f2bf(p1 * inv);
      prow[32] = f2bf(p2 * inv);
      prow[48] = f2bf(p3 * inv);
    }
  }
  __syncthreads();

  // ---- Phase 3: O = P @ V -> O row-major [token][dim] bf16 (aliases Q) ----
  for (int i = 0; i < 4; ++i) {
    int t = wid + 8 * i;
    int hd = t >> 3, mt = (t >> 1) & 3, nt = t & 1;
    floatx8 acc = {};
#pragma unroll
    for (int kk = 0; kk < 2; ++kk) {
      bf16x16 A = ldA(Pm + hd * (64 * LDP), LDP, mt * 16, kk * 32, lane);
      bf16x16 B = ldB(Vt, LDV, hd * HD + nt * 16, kk * 32, lane);   // Vt rows = dims
      acc = wmma_bf16(A, B, acc);
    }
    int dim = hd * HD + nt * 16 + r16;
#pragma unroll
    for (int r = 0; r < 8; ++r)
      Om[(mt * 16 + 8 * hh + r) * LDX + dim] = f2bf(acc[r]);
  }
  __syncthreads();

  // ---- Phase 4: out^T = Wp @ O^T + b -> contiguous fp32 stores to HBM ----
  for (int i = 0; i < 4; ++i) {
    int t = wid + 8 * i;
    int mt = t >> 2, nt = t & 3;
    floatx8 acc = {};
#pragma unroll
    for (int kk = 0; kk < 4; ++kk) {
      bf16x16 A = ldA(projw, DIMC, mt * 16, kk * 32, lane);
      bf16x16 B = ldB(Om, LDX, nt * 16, kk * 32, lane);
      acc = wmma_bf16(A, B, acc);
    }
    int token = nt * 16 + r16;
    if (token < NTOK) {
      int feat0 = mt * 16 + 8 * hh;
      float* dst = out + ((size_t)blk * NTOK + token) * DIMC + feat0;
      float4 lo = make_float4(acc[0] + proj_b[feat0 + 0], acc[1] + proj_b[feat0 + 1],
                              acc[2] + proj_b[feat0 + 2], acc[3] + proj_b[feat0 + 3]);
      float4 hi = make_float4(acc[4] + proj_b[feat0 + 4], acc[5] + proj_b[feat0 + 5],
                              acc[6] + proj_b[feat0 + 6], acc[7] + proj_b[feat0 + 7]);
      *(float4*)dst = lo;
      *(float4*)(dst + 4) = hi;
    }
  }
}

extern "C" void kernel_launch(void* const* d_in, const int* in_sizes, int n_in,
                              void* d_out, int out_size, void* d_ws, size_t ws_size,
                              hipStream_t stream) {
  const float* x          = (const float*)d_in[0];
  const float* mask       = (const float*)d_in[1];
  const float* qkv_w      = (const float*)d_in[2];
  const float* qkv_b      = (const float*)d_in[3];
  const float* proj_w     = (const float*)d_in[4];
  const float* proj_b     = (const float*)d_in[5];
  const float* bias_table = (const float*)d_in[6];
  const int*   rel_index  = (const int*)d_in[7];
  float* out = (float*)d_out;

  u16*   qkvw_bf  = (u16*)d_ws;                              // 384*128 bf16 = 98304 B
  u16*   projw_bf = (u16*)((char*)d_ws + 98304);             // 128*128 bf16 = 32768 B
  float* biasPre  = (float*)((char*)d_ws + 131072);          // 4*49*49 f32 = 38416 B

  const int nprep = 384 * DIMC + DIMC * DIMC + NH * NTOK * NTOK;
  swin_prep_kernel<<<(nprep + 255) / 256, 256, 0, stream>>>(
      qkv_w, proj_w, bias_table, rel_index, qkvw_bf, projw_bf, biasPre);

  const int B = in_sizes[0] / (NTOK * DIMC);                 // 4096 windows
  swin_attn_kernel<<<B, 256, 0, stream>>>(
      x, mask, qkv_b, proj_b, qkvw_bf, projw_bf, biasPre, out);
}
